// Attention_90228672954441
// MI455X (gfx1250) — matrix operand order repaired and checked by reading, not compiled, not run on hardware
//
#include <hip/hip_runtime.h>
#include <hip/hip_bf16.h>

// ---------------------------------------------------------------- types
typedef __attribute__((ext_vector_type(8)))  _Float16 v8h;
typedef __attribute__((ext_vector_type(16))) _Float16 v16h;
typedef __attribute__((ext_vector_type(8)))  float    v8f;
typedef __attribute__((ext_vector_type(4)))  unsigned int u32x4;
typedef __attribute__((ext_vector_type(8)))  int      i32x8;
typedef __attribute__((ext_vector_type(4)))  int      i32x4;

union V16u { v16h v; v8h h[2]; };

// ---------------------------------------------------------------- constants
constexpr int kB = 2, kN = 4096, kC = 256, kH = 8, kD = 32, kM = 1024;

#if defined(__has_builtin)
#if __has_builtin(__builtin_amdgcn_tensor_load_to_lds) && defined(__AMDGCN__)
#define USE_TDM 1
#endif
#endif

// ---------------------------------------------------------------- tile loaders
// A-matrix 16x32 f16 layout: lanes 0-15 hold row M=lane with
// K=0..15; lanes 16-31 hold row M=lane-16 with K=16..31 (matches B).
__device__ __forceinline__ v16h load_A_tile(const _Float16* __restrict__ base,
                                            int lda, int lane) {
  int row = lane & 15;
  int kb  = (lane >> 4) * 16;
  const _Float16* p = base + (size_t)row * lda + kb;
  V16u u;
  u.h[0] = *(const v8h*)(p);
  u.h[1] = *(const v8h*)(p + 8);
  return u.v;
}
__device__ __forceinline__ v16h load_A_tile_lds(const _Float16* base, int lda, int lane) {
  int row = lane & 15;
  int kb  = (lane >> 4) * 16;
  const _Float16* p = base + row * lda + kb;
  V16u u;
  u.h[0] = *(const v8h*)(p);
  u.h[1] = *(const v8h*)(p + 8);
  return u.v;
}
// B-matrix 32x16 f16 layout: lanes 0-15 hold column N=lane, K=0..15;
// lanes 16-31 hold column N=lane-16, K=16..31. One contiguous 16-half run.
__device__ __forceinline__ v16h load_B_tile(const _Float16* __restrict__ base,
                                            int ldb, int lane) {
  int col = lane & 15;
  int kb  = (lane >> 4) * 16;
  const _Float16* p = base + (size_t)col * ldb + kb;
  V16u u;
  u.h[0] = *(const v8h*)(p);
  u.h[1] = *(const v8h*)(p + 8);
  return u.v;
}
__device__ __forceinline__ v16h load_B_tile_lds(const _Float16* base, int ldb, int lane) {
  int col = lane & 15;
  int kb  = (lane >> 4) * 16;
  const _Float16* p = base + col * ldb + kb;
  V16u u;
  u.h[0] = *(const v8h*)(p);
  u.h[1] = *(const v8h*)(p + 8);
  return u.v;
}

// ---------------------------------------------------------------- TDM helper
#ifdef USE_TDM
// 1-D contiguous copy: n_qwords 8-byte elements, global -> LDS offset.
// D# per CDNA5 ISA ch.8: group0 = {flags, lds_addr, global_addr, type=2},
// group1 = {data_size=3(8B), tensor_dim0=tile_dim0=n, stride0=n}.
// Toolchain here is the 6-arg form: (g0, g1, g2, g3, g4, cpol).
__device__ __forceinline__ void tdm_load_1d(unsigned lds_off, const void* gptr,
                                            unsigned n_qwords) {
  unsigned long long ga = (unsigned long long)gptr;
  u32x4 g0;
  g0[0] = 1u;                                             // count=1, user mode
  g0[1] = lds_off;                                        // lds_addr (bytes)
  g0[2] = (unsigned)(ga & 0xffffffffu);                   // global_addr[31:0]
  g0[3] = (unsigned)((ga >> 32) & 0x01ffffffu) | (2u << 30); // [56:32] | type=2
  i32x8 g1;
  g1[0] = (3 << 16);                                      // data_size=3 (8B), mask=0
  g1[1] = (int)((n_qwords & 0xffffu) << 16);              // tensor_dim0[15:0]
  g1[2] = (int)((n_qwords >> 16) & 0xffffu) | (1 << 16);  // tensor_dim0[31:16], tensor_dim1=1
  g1[3] = (int)((n_qwords & 0xffffu) << 16);              // tile_dim0 = n
  g1[4] = 0;                                              // tile_dim1/2 unused
  g1[5] = (int)n_qwords;                                  // tensor_dim0_stride[31:0]
  g1[6] = 0;
  g1[7] = 0;
  i32x4 z4 = {0, 0, 0, 0};
  i32x8 z8 = {0, 0, 0, 0, 0, 0, 0, 0};
  __builtin_amdgcn_tensor_load_to_lds(g0, g1, z4, z4, z8, 0);
}
#endif

// ---------------------------------------------------------------- conversions
__global__ void cvt_f32_f16(const float* __restrict__ in, _Float16* __restrict__ out, int n) {
  int i = blockIdx.x * blockDim.x + threadIdx.x;
  if (i < n) out[i] = (_Float16)in[i];
}

// Patch extraction: p[b, m, j] with j = i*4 + dy*2 + dx; m over 32x32 output grid.
__global__ void build_patches(const float* __restrict__ x, _Float16* __restrict__ p) {
  int idx = blockIdx.x * blockDim.x + threadIdx.x;   // B*M*1024
  if (idx >= kB * kM * 1024) return;
  int j  = idx & 1023;
  int bm = idx >> 10;
  int m  = bm & (kM - 1);
  int b  = bm >> 10;
  int i  = j >> 2, dy = (j >> 1) & 1, dx = j & 1;
  int y  = m >> 5, xc = m & 31;
  int n  = (2 * y + dy) * 64 + (2 * xc + dx);
  p[idx] = (_Float16)x[((size_t)(b * kN + n)) * kC + i];
}

// ---------------------------------------------------------------- generic WMMA GEMM
// Y[r, o] = (sum_k A16[r,k] * W16[o,k]) * scale + bias[o]
// grid = (R/16, O/256), block = 128 (4 waves); each wave computes a 16x64 tile
// with 4 accumulators (A reused 4x) and software double-buffered tile loads.
__global__ void __launch_bounds__(128)
gemm_wmma(const _Float16* __restrict__ A, const _Float16* __restrict__ W,
          const float* __restrict__ bias, void* __restrict__ Y,
          int K, int ldY, int out_f32, float out_scale) {
  int lane  = threadIdx.x & 31;
  int wave  = threadIdx.x >> 5;
  int rBase = blockIdx.x * 16;
  int oBase = (blockIdx.y * 4 + wave) * 64;
  const _Float16* a0 = A + (size_t)rBase * K;
  const _Float16* w0 = W + (size_t)oBase * K;
  const _Float16* w1 = w0 + (size_t)16 * K;
  const _Float16* w2 = w0 + (size_t)32 * K;
  const _Float16* w3 = w0 + (size_t)48 * K;

  v8f acc0 = {}, acc1 = {}, acc2 = {}, acc3 = {};
  v16h a  = load_A_tile(a0, K, lane);
  v16h b0 = load_B_tile(w0, K, lane);
  v16h b1 = load_B_tile(w1, K, lane);
  v16h b2 = load_B_tile(w2, K, lane);
  v16h b3 = load_B_tile(w3, K, lane);

  int steps = K >> 5;
  for (int t = 1; t < steps; ++t) {
    int kk = t * 32;
    // speculative prefetch one chunk ahead (dropped safely if out of range)
    __builtin_prefetch(a0 + (size_t)(lane & 15) * K + kk + 32, 0, 1);
    v16h an = load_A_tile(a0 + kk, K, lane);
    v16h c0 = load_B_tile(w0 + kk, K, lane);
    v16h c1 = load_B_tile(w1 + kk, K, lane);
    v16h c2 = load_B_tile(w2 + kk, K, lane);
    v16h c3 = load_B_tile(w3 + kk, K, lane);
    acc0 = __builtin_amdgcn_wmma_f32_16x16x32_f16(false, a, false, b0, (short)0, acc0, false, false);
    acc1 = __builtin_amdgcn_wmma_f32_16x16x32_f16(false, a, false, b1, (short)0, acc1, false, false);
    acc2 = __builtin_amdgcn_wmma_f32_16x16x32_f16(false, a, false, b2, (short)0, acc2, false, false);
    acc3 = __builtin_amdgcn_wmma_f32_16x16x32_f16(false, a, false, b3, (short)0, acc3, false, false);
    a = an; b0 = c0; b1 = c1; b2 = c2; b3 = c3;
  }
  acc0 = __builtin_amdgcn_wmma_f32_16x16x32_f16(false, a, false, b0, (short)0, acc0, false, false);
  acc1 = __builtin_amdgcn_wmma_f32_16x16x32_f16(false, a, false, b1, (short)0, acc1, false, false);
  acc2 = __builtin_amdgcn_wmma_f32_16x16x32_f16(false, a, false, b2, (short)0, acc2, false, false);
  acc3 = __builtin_amdgcn_wmma_f32_16x16x32_f16(false, a, false, b3, (short)0, acc3, false, false);

  int hs = lane >> 4;
  v8f* accs[4] = {&acc0, &acc1, &acc2, &acc3};
#pragma unroll
  for (int tile = 0; tile < 4; ++tile) {
    int   col = oBase + tile * 16 + (lane & 15);
    float bv  = bias ? bias[col] : 0.0f;
#pragma unroll
    for (int g = 0; g < 8; ++g) {
      int   row = rBase + g + 8 * hs;
      float val = (*accs[tile])[g] * out_scale + bv;
      if (out_f32) ((float*)Y)[(size_t)row * ldY + col] = val;
      else         ((_Float16*)Y)[(size_t)row * ldY + col] = (_Float16)val;
    }
  }
}

// ---------------------------------------------------------------- layernorm (one wave/row, C=256)
__global__ void layernorm_rows(const float* __restrict__ xs, const float* __restrict__ w,
                               const float* __restrict__ bb, _Float16* __restrict__ out,
                               int rows) {
  int gwave = (blockIdx.x * blockDim.x + threadIdx.x) >> 5;
  int lane  = threadIdx.x & 31;
  if (gwave >= rows) return;
  const float* r = xs + (size_t)gwave * kC;
  float v[8], s = 0.f, ss = 0.f;
#pragma unroll
  for (int t = 0; t < 8; ++t) { v[t] = r[lane + 32 * t]; s += v[t]; ss += v[t] * v[t]; }
#pragma unroll
  for (int m = 16; m >= 1; m >>= 1) { s += __shfl_xor(s, m, 32); ss += __shfl_xor(ss, m, 32); }
  float mu  = s * (1.0f / kC);
  float var = ss * (1.0f / kC) - mu * mu;
  float inv = rsqrtf(var + 1e-5f);
  _Float16* o = out + (size_t)gwave * kC;
#pragma unroll
  for (int t = 0; t < 8; ++t) {
    int c = lane + 32 * t;
    o[c] = (_Float16)((v[t] - mu) * inv * w[c] + bb[c]);
  }
}

// ---------------------------------------------------------------- kv scatter
__global__ void scatter_kv(const _Float16* __restrict__ kv, _Float16* __restrict__ k,
                           _Float16* __restrict__ vT) {
  int idx = blockIdx.x * blockDim.x + threadIdx.x;   // B*H*M*D
  if (idx >= kB * kH * kM * kD) return;
  int d  = idx & 31;
  int m  = (idx >> 5) & (kM - 1);
  int hh = (idx >> 15) & 7;
  int b  = idx >> 18;
  size_t src = ((size_t)(b * kM + m)) * (2 * kC) + hh * kD + d;
  k[idx] = kv[src];
  vT[(((size_t)(b * kH + hh) * kD + d) * kM) + m] = kv[src + kC];
}

// ---------------------------------------------------------------- flash attention
// grid = (B*H, N/256), block = 512 (16 waves, 16 q-rows each).
// Whole head K (1024x32) and V^T (32x1024) staged into LDS via the Tensor Data
// Mover (TENSOR_LOAD_TO_LDS + s_wait_tensorcnt), then all WMMA B-tiles come
// from LDS. d=32 matches WMMA K exactly; softmax scale pre-folded into q.
__global__ void __launch_bounds__(512)
attention_wmma(const _Float16* __restrict__ q, const _Float16* __restrict__ k,
               const _Float16* __restrict__ vT, _Float16* __restrict__ o) {
  extern __shared__ _Float16 smem[];
  _Float16* kL = smem;              // 1024 x 32 f16 : 64 KB  (lds offset 0)
  _Float16* vL = smem + 32768;      // 32 x 1024 f16 : 64 KB  (lds offset 65536)
  _Float16* pL = smem + 65536;      // 16 waves x 16x32 P tiles : 16 KB

  int lane = threadIdx.x & 31;
  int wave = threadIdx.x >> 5;
  int bh   = blockIdx.x;
  int b    = bh >> 3, hh = bh & 7;
  int hs   = lane >> 4;

  const _Float16* kG = k  + (size_t)bh * kM * kD;
  const _Float16* vG = vT + (size_t)bh * kD * kM;

#ifdef USE_TDM
  if (wave == 0) {
    tdm_load_1d(0u,     kG, (kM * kD) / 4);   // 8192 qwords = 64 KB
    tdm_load_1d(65536u, vG, (kD * kM) / 4);
    __builtin_amdgcn_s_wait_tensorcnt(0);
  }
  __syncthreads();
#else
  for (int i = threadIdx.x; i < (kM * kD) / 8; i += blockDim.x) {
    ((v8h*)kL)[i] = ((const v8h*)kG)[i];
    ((v8h*)vL)[i] = ((const v8h*)vG)[i];
  }
  __syncthreads();
#endif

  int nBase = blockIdx.y * 256 + wave * 16;
  const _Float16* qbase = q + ((size_t)(b * kN + nBase)) * kC + hh * kD;
  v16h qa = load_A_tile(qbase, kC, lane);

  v8f o0 = {}, o1 = {};
  float mi[8], li[8];
#pragma unroll
  for (int g = 0; g < 8; ++g) { mi[g] = -1e30f; li[g] = 0.0f; }

  _Float16* lt = pL + wave * 512;
  int cm = lane & 15;

  for (int mc = 0; mc < kM; mc += 32) {
    v16h kb0 = load_B_tile_lds(kL + mc * kD, kD, lane);
    v16h kb1 = load_B_tile_lds(kL + (mc + 16) * kD, kD, lane);
    v8f z = {};
    v8f s0 = __builtin_amdgcn_wmma_f32_16x16x32_f16(false, qa, false, kb0,
                                                    (short)0, z, false, false);
    v8f s1 = __builtin_amdgcn_wmma_f32_16x16x32_f16(false, qa, false, kb1,
                                                    (short)0, z, false, false);
    float p0[8], p1[8];
#pragma unroll
    for (int g = 0; g < 8; ++g) {
      float mx = fmaxf(s0[g], s1[g]);
#pragma unroll
      for (int msk = 8; msk >= 1; msk >>= 1) mx = fmaxf(mx, __shfl_xor(mx, msk, 16));
      float nm    = fmaxf(mi[g], mx);
      float alpha = __expf(mi[g] - nm);
      float e0    = __expf(s0[g] - nm);
      float e1    = __expf(s1[g] - nm);
      float rs    = e0 + e1;
#pragma unroll
      for (int msk = 8; msk >= 1; msk >>= 1) rs += __shfl_xor(rs, msk, 16);
      li[g] = li[g] * alpha + rs;
      mi[g] = nm;
      o0[g] *= alpha; o1[g] *= alpha;
      p0[g] = e0; p1[g] = e1;
    }
    // D-layout -> A-layout transpose through wave-private LDS tile (f16)
#pragma unroll
    for (int g = 0; g < 8; ++g) {
      int row = g + 8 * hs;
      lt[row * 32 + cm]      = (_Float16)p0[g];
      lt[row * 32 + 16 + cm] = (_Float16)p1[g];
    }
    v16h pa  = load_A_tile_lds(lt, 32, lane);
    v16h vb0 = load_B_tile_lds(vL + mc, kM, lane);
    v16h vb1 = load_B_tile_lds(vL + 16 * kM + mc, kM, lane);
    o0 = __builtin_amdgcn_wmma_f32_16x16x32_f16(false, pa, false, vb0,
                                                (short)0, o0, false, false);
    o1 = __builtin_amdgcn_wmma_f32_16x16x32_f16(false, pa, false, vb1,
                                                (short)0, o1, false, false);
  }
#pragma unroll
  for (int g = 0; g < 8; ++g) {
    float inv = 1.0f / li[g];
    int   n   = nBase + g + 8 * hs;
    size_t ro = ((size_t)(b * kN + n)) * kC + hh * kD;
    o[ro + cm]      = (_Float16)(o0[g] * inv);
    o[ro + 16 + cm] = (_Float16)(o1[g] * inv);
  }
}

// ---------------------------------------------------------------- host launch
extern "C" void kernel_launch(void* const* d_in, const int* in_sizes, int n_in,
                              void* d_out, int out_size, void* d_ws, size_t ws_size,
                              hipStream_t stream) {
  const float* xin[2]  = {(const float*)d_in[0], (const float*)d_in[1]};
  const float* Wq      = (const float*)d_in[2];
  const float* Wkv     = (const float*)d_in[3];
  const float* Wproj   = (const float*)d_in[4];
  const float* bproj   = (const float*)d_in[5];
  const float* Wsr     = (const float*)d_in[6];
  const float* bsr     = (const float*)d_in[7];
  const float* lnw[2]  = {(const float*)d_in[8],  (const float*)d_in[10]};
  const float* lnb[2]  = {(const float*)d_in[9],  (const float*)d_in[11]};
  float*       out     = (float*)d_out;

  char* ws = (char*)d_ws;
  _Float16* Wq16    = (_Float16*)(ws);                 // 128 KB
  _Float16* Wkv16   = (_Float16*)(ws + (1u << 17));    // 256 KB
  _Float16* Wproj16 = (_Float16*)(ws + 393216);        // 128 KB
  _Float16* Wsr16   = (_Float16*)(ws + 524288);        // 512 KB
  _Float16* x16     = (_Float16*)(ws + 1048576);       // 4 MB
  _Float16* p16     = (_Float16*)(ws + 5242880);       // 4 MB
  float*    xs32    = (float*)   (ws + 9437184);       // 2 MB
  _Float16* xs16    = (_Float16*)(ws + 11534336);      // 1 MB
  _Float16* kv16    = (_Float16*)(ws + 12582912);      // 2 MB
  _Float16* k16     = (_Float16*)(ws + 14680064);      // 1 MB
  _Float16* vT16    = (_Float16*)(ws + 15728640);      // 1 MB
  _Float16* q16     = (_Float16*)(ws + 16777216);      // 4 MB
  _Float16* o16     = (_Float16*)(ws + 20971520);      // 4 MB  (end 24 MB)

  const int TPB = 256;
  auto blocks = [](int n) { return (n + 255) / 256; };

  // weights -> f16 (once)
  cvt_f32_f16<<<blocks(kC * kC),       TPB, 0, stream>>>(Wq,    Wq16,    kC * kC);
  cvt_f32_f16<<<blocks(2 * kC * kC),   TPB, 0, stream>>>(Wkv,   Wkv16,   2 * kC * kC);
  cvt_f32_f16<<<blocks(kC * kC),       TPB, 0, stream>>>(Wproj, Wproj16, kC * kC);
  cvt_f32_f16<<<blocks(kC * kC * 4),   TPB, 0, stream>>>(Wsr,   Wsr16,   kC * kC * 4);

  const float qscale = 0.17677669529663689f;   // 1/sqrt(32)

  for (int s = 0; s < 2; ++s) {
    cvt_f32_f16<<<blocks(kB * kN * kC), TPB, 0, stream>>>(xin[s], x16, kB * kN * kC);
    build_patches<<<blocks(kB * kM * 1024), TPB, 0, stream>>>(xin[s], p16);

    // conv-as-GEMM: (2048 x 1024) @ (256 x 1024)^T + bsr -> f32
    gemm_wmma<<<dim3(kB * kM / 16, 1), 128, 0, stream>>>(
        p16, Wsr16, bsr, xs32, 1024, kC, 1, 1.0f);

    layernorm_rows<<<kB * kM / 8, 256, 0, stream>>>(xs32, lnw[s], lnb[s], xs16, kB * kM);

    // kv: (2048 x 256) @ (512 x 256)^T -> f16
    gemm_wmma<<<dim3(kB * kM / 16, 2), 128, 0, stream>>>(
        xs16, Wkv16, nullptr, kv16, kC, 2 * kC, 0, 1.0f);
    scatter_kv<<<blocks(kB * kH * kM * kD), TPB, 0, stream>>>(kv16, k16, vT16);

    // q: (8192 x 256) @ (256 x 256)^T, scaled by 1/sqrt(d) -> f16
    gemm_wmma<<<dim3(kB * kN / 16, 1), 128, 0, stream>>>(
        x16, Wq16, nullptr, q16, kC, kC, 0, qscale);

    // attention: TDM-staged K/V^T in LDS (144 KB dynamic)
    attention_wmma<<<dim3(kB * kH, kN / 256), 512, 147456, stream>>>(q16, k16, vT16, o16);

    // output projection -> f32 into d_out slice
    gemm_wmma<<<dim3(kB * kN / 16, 1), 128, 0, stream>>>(
        o16, Wproj16, bproj, out + (size_t)s * kB * kN * kC, kC, kC, 1, 1.0f);
  }
  (void)in_sizes; (void)n_in; (void)out_size; (void)ws_size;
}
